// FeatureClassifier_61589831024754
// MI455X (gfx1250) — compile-verified, matrix operand order checked by reading, not run on hardware
//
#include <hip/hip_runtime.h>
#include <cstdint>
#include <cstddef>

typedef __attribute__((ext_vector_type(16))) _Float16 v16h;
typedef __attribute__((ext_vector_type(8)))  float    v8f;

#define BN_SCALE 0.99999500003749969f  /* 1/sqrt(1+1e-5) */

__device__ __forceinline__ float selu_f(float v) {
    const float S = 1.0507009873554805f, A = 1.6732632423543772f;
    return v > 0.f ? S * v : S * A * (expf(v) - 1.f);
}

// ---------------------------------------------------------------------------
// Stage 0: reductions of e (B, D, NS, NT)
// ---------------------------------------------------------------------------
__global__ void k_reduce_eS(const float* __restrict__ e, const float* __restrict__ pos,
                            float* __restrict__ eS, int B, int D, int NS, int NT) {
    int idx = blockIdx.x * blockDim.x + threadIdx.x;
    if (idx >= B * NS * D) return;
    int d = idx % D; int s = (idx / D) % NS; int b = idx / (D * NS);
    const float* p = e + (((size_t)b * D + d) * NS + s) * NT;
    float m = 0.f;
    for (int t = 0; t < NT; ++t) m = fmaxf(m, fabsf(p[t]));
    eS[idx] = m + pos[s * D + d];
}

__global__ void k_reduce_eT(const float* __restrict__ e, float* __restrict__ eT,
                            int B, int D, int NS, int NT) {
    int idx = blockIdx.x * blockDim.x + threadIdx.x;
    if (idx >= B * NT * D) return;
    int d = idx % D; int t = (idx / D) % NT; int b = idx / (D * NT);
    const float* p = e + (((size_t)b * D + d) * NS) * (size_t)NT + t;
    float m = 0.f;
    for (int s = 0; s < NS; ++s) m = fmaxf(m, fabsf(p[(size_t)s * NT]));
    eT[idx] = m;
}

// ---------------------------------------------------------------------------
// Pairwise attention scores via WMMA (split-f16, fp32 accumulate).
// scores[b,i,j] = tanh((x_i .* x_j) @ W + bias) . aw_sel(i,j)  * inv_temp
// aw_sel: a11 if (i<n1 && j<n1), a22 if (i>=n1 && j>=n1), else a12.
// One wave per (jtile, i, b). DIN in {32,64}, DOUT in {32,64}.
// Fragment mapping per CDNA5 ISA 7.12.2:
//   A (16x32 f16): lane L -> M=L%16; elem h -> K = (h/8)*16 + (L>=16?8:0) + h%8
//   B (32x16 f16): lane L -> N=L%16; same K pattern
//   C (16x16 f32): vgpr r, lane L -> M = r + 8*(L/16), N = L%16
// ---------------------------------------------------------------------------
template<int DIN, int DOUT>
__global__ __launch_bounds__(32) void k_pair_att(
    const float* __restrict__ x, const float* __restrict__ W, const float* __restrict__ bias,
    const float* __restrict__ aw11, const float* __restrict__ aw22, const float* __restrict__ aw12,
    float* __restrict__ scores, int n, int n1, float inv_temp)
{
    constexpr int KSTEPS = DIN / 32;
    constexpr int NTILES = DOUT / 16;
    int jt = blockIdx.x, i = blockIdx.y, b = blockIdx.z;
    int lane = threadIdx.x & 31;
    int half = lane >> 4;
    int l16  = lane & 15;
    int j  = jt * 16 + l16;
    int jc = j < n ? j : n - 1;
    float jmask = j < n ? 1.f : 0.f;
    const float* xi = x + ((size_t)b * n + i)  * DIN;
    const float* xj = x + ((size_t)b * n + jc) * DIN;

    v16h aHi[KSTEPS], aLo[KSTEPS];
#pragma unroll
    for (int ks = 0; ks < KSTEPS; ++ks) {
#pragma unroll
        for (int h = 0; h < 16; ++h) {
            int k = ks * 32 + ((h >> 3) << 4) + (half << 3) + (h & 7);
            float a = xi[k] * xj[k] * jmask;
            _Float16 hi = (_Float16)a;
            aHi[ks][h] = hi;
            aLo[ks][h] = (_Float16)(a - (float)hi);
        }
    }

    bool rowi = (i < n1);
    float part[8];
#pragma unroll
    for (int r = 0; r < 8; ++r) part[r] = 0.f;

#pragma unroll
    for (int nt = 0; nt < NTILES; ++nt) {
        int o = nt * 16 + l16;
        v8f c = {};
#pragma unroll
        for (int ks = 0; ks < KSTEPS; ++ks) {
            v16h bHi, bLo;
#pragma unroll
            for (int h = 0; h < 16; ++h) {
                int k = ks * 32 + ((h >> 3) << 4) + (half << 3) + (h & 7);
                float w = W[k * DOUT + o];
                _Float16 hi = (_Float16)w;
                bHi[h] = hi;
                bLo[h] = (_Float16)(w - (float)hi);
            }
            c = __builtin_amdgcn_wmma_f32_16x16x32_f16(false, aHi[ks], false, bHi, (short)0, c, false, false);
            c = __builtin_amdgcn_wmma_f32_16x16x32_f16(false, aHi[ks], false, bLo, (short)0, c, false, false);
            c = __builtin_amdgcn_wmma_f32_16x16x32_f16(false, aLo[ks], false, bHi, (short)0, c, false, false);
        }
        float bo  = bias[o];
        float w11 = aw11[o], w22 = aw22[o], w12 = aw12[o];
#pragma unroll
        for (int r = 0; r < 8; ++r) {
            int jr = jt * 16 + r + 8 * half;
            bool colj = (jr < n1);
            float wsel = (rowi && colj) ? w11 : ((!rowi && !colj) ? w22 : w12);
            part[r] += tanhf(c[r] + bo) * wsel;
        }
    }
    // reduce over the 16 columns held by each 16-lane half
#pragma unroll
    for (int r = 0; r < 8; ++r) {
        float v = part[r];
        v += __shfl_xor(v, 1, 32);
        v += __shfl_xor(v, 2, 32);
        v += __shfl_xor(v, 4, 32);
        v += __shfl_xor(v, 8, 32);
        part[r] = v;
    }
    if (l16 == 0) {
#pragma unroll
        for (int r = 0; r < 8; ++r) {
            int jr = jt * 16 + r + 8 * half;
            if (jr < n) scores[((size_t)b * n + i) * n + jr] = part[r] * inv_temp;
        }
    }
}

// ---------------------------------------------------------------------------
// In-place row softmax: one 256-thread block per row of length L.
// ---------------------------------------------------------------------------
__global__ void k_softmax_rows(float* __restrict__ a, int L) {
    __shared__ float red[256];
    float* row = a + (size_t)blockIdx.x * L;
    int tid = threadIdx.x;
    float m = -3.0e38f;
    for (int j = tid; j < L; j += 256) m = fmaxf(m, row[j]);
    red[tid] = m; __syncthreads();
    for (int s = 128; s > 0; s >>= 1) { if (tid < s) red[tid] = fmaxf(red[tid], red[tid + s]); __syncthreads(); }
    m = red[0]; __syncthreads();
    float acc = 0.f;
    for (int j = tid; j < L; j += 256) { float e2 = expf(row[j] - m); row[j] = e2; acc += e2; }
    red[tid] = acc; __syncthreads();
    for (int s = 128; s > 0; s >>= 1) { if (tid < s) red[tid] += red[tid + s]; __syncthreads(); }
    float inv = 1.f / red[0];
    for (int j = tid; j < L; j += 256) row[j] *= inv;
}

// out[b,i,d] = sum_j wts[b,i,j] * x[b,j,d]     (ni rows of weights per batch)
__global__ void k_attn_agg(const float* __restrict__ wts, const float* __restrict__ x,
                           float* __restrict__ out, int B, int ni, int n, int d) {
    int idx = blockIdx.x * blockDim.x + threadIdx.x;
    if (idx >= B * ni * d) return;
    int o = idx % d; int i = (idx / d) % ni; int b = idx / (d * ni);
    const float* wr = wts + ((size_t)b * ni + i) * n;
    const float* xb = x + (size_t)b * n * d;
    float acc = 0.f;
    for (int j = 0; j < n; ++j) acc += wr[j] * xb[(size_t)j * d + o];
    out[idx] = acc;
}

// out[b, roff+i, o] = in[b,i,:] @ W + bias   (strided batches so we can concat)
__global__ void k_linear(const float* __restrict__ in, const float* __restrict__ W,
                         const float* __restrict__ bias, float* __restrict__ out,
                         int B, int n, int din, int dout,
                         int in_bstride, int out_bstride, int out_roff) {
    int idx = blockIdx.x * blockDim.x + threadIdx.x;
    if (idx >= B * n * dout) return;
    int o = idx % dout; int i = (idx / dout) % n; int b = idx / (dout * n);
    const float* xp = in + (size_t)b * in_bstride + (size_t)i * din;
    float acc = bias[o];
    for (int d = 0; d < din; ++d) acc += xp[d] * W[d * dout + o];
    out[(size_t)b * out_bstride + (size_t)(out_roff + i) * dout + o] = acc;
}

// h = selu((agg@W1 + b1 + x@W2 + b2) * BN_SCALE)
__global__ void k_proj_selu(const float* __restrict__ agg, const float* __restrict__ x,
                            const float* __restrict__ W1, const float* __restrict__ b1,
                            const float* __restrict__ W2, const float* __restrict__ b2,
                            float* __restrict__ out, int rows, int din, int dout) {
    int idx = blockIdx.x * blockDim.x + threadIdx.x;
    if (idx >= rows * dout) return;
    int o = idx % dout; int r = idx / dout;
    const float* ap = agg + (size_t)r * din;
    const float* xp = x   + (size_t)r * din;
    float acc = b1[o] + b2[o];
    for (int d = 0; d < din; ++d) acc += ap[d] * W1[d * dout + o] + xp[d] * W2[d * dout + o];
    out[idx] = selu_f(acc * BN_SCALE);
}

// sM[b,i] = tanh((x_i .* master_b) @ W + bias) . aw * inv_temp
__global__ void k_master_att(const float* __restrict__ x, const float* __restrict__ master, int mstride,
                             const float* __restrict__ W, const float* __restrict__ bias,
                             const float* __restrict__ aw, float* __restrict__ out,
                             int B, int n, int din, int dout, float inv_temp) {
    int idx = blockIdx.x * blockDim.x + threadIdx.x;
    if (idx >= B * n) return;
    int i = idx % n; int b = idx / n;
    const float* xi = x + ((size_t)b * n + i) * din;
    const float* mb = master + (size_t)b * mstride;
    float s = 0.f;
    for (int o = 0; o < dout; ++o) {
        float acc = bias[o];
        for (int d = 0; d < din; ++d) acc += xi[d] * mb[d] * W[d * dout + o];
        s += tanhf(acc) * aw[o];
    }
    out[idx] = s * inv_temp;
}

// new_master[b,o] = magg[b]@W1 + b1 + master[b]@W2 + b2
__global__ void k_master_update(const float* __restrict__ magg, const float* __restrict__ master, int mstride,
                                const float* __restrict__ W1, const float* __restrict__ b1,
                                const float* __restrict__ W2, const float* __restrict__ b2,
                                float* __restrict__ out, int B, int din, int dout) {
    int idx = blockIdx.x * blockDim.x + threadIdx.x;
    if (idx >= B * dout) return;
    int o = idx % dout; int b = idx / dout;
    float acc = b1[o] + b2[o];
    for (int d = 0; d < din; ++d)
        acc += magg[(size_t)b * din + d] * W1[d * dout + o] + master[(size_t)b * mstride + d] * W2[d * dout + o];
    out[idx] = acc;
}

// sigmoid-score top-k pool (torch.topk semantics: desc values, ties -> lower index)
__global__ void k_graph_pool(const float* __restrict__ h, const float* __restrict__ w,
                             const float* __restrict__ bb, float* __restrict__ out,
                             int n, int d, int k) {
    __shared__ float sc[512];
    __shared__ float sc0[512];
    __shared__ int   sel[256];
    __shared__ float rv[256];
    __shared__ int   ri[256];
    int b = blockIdx.x, tid = threadIdx.x;
    const float* hb = h + (size_t)b * n * d;
    for (int i = tid; i < n; i += 256) {
        float acc = bb[0];
        for (int dd = 0; dd < d; ++dd) acc += hb[(size_t)i * d + dd] * w[dd];
        float s = 1.f / (1.f + expf(-acc));
        sc[i] = s; sc0[i] = s;
    }
    __syncthreads();
    for (int r = 0; r < k; ++r) {
        float bv = -1e30f; int bi = 0x7fffffff;
        for (int i = tid; i < n; i += 256) {
            float v = sc[i];
            if (v > bv) { bv = v; bi = i; }
        }
        rv[tid] = bv; ri[tid] = bi; __syncthreads();
        for (int s2 = 128; s2 > 0; s2 >>= 1) {
            if (tid < s2) {
                if (rv[tid + s2] > rv[tid] || (rv[tid + s2] == rv[tid] && ri[tid + s2] < ri[tid])) {
                    rv[tid] = rv[tid + s2]; ri[tid] = ri[tid + s2];
                }
            }
            __syncthreads();
        }
        if (tid == 0) { sel[r] = ri[0]; sc[ri[0]] = -1e30f; }
        __syncthreads();
    }
    for (int t2 = tid; t2 < k * d; t2 += 256) {
        int r = t2 / d, dd = t2 % d; int i = sel[r];
        out[((size_t)b * k + r) * d + dd] = hb[(size_t)i * d + dd] * sc0[i];
    }
}

__global__ void k_copy_rows(const float* __restrict__ src, float* __restrict__ dst,
                            int B, int nsrc, int d, int roff, int nrows) {
    int idx = blockIdx.x * blockDim.x + threadIdx.x;
    if (idx >= B * nrows * d) return;
    int dd = idx % d; int i = (idx / d) % nrows; int b = idx / (d * nrows);
    dst[idx] = src[((size_t)b * nsrc + roff + i) * d + dd];
}

__global__ void k_add_rows(float* __restrict__ dst, const float* __restrict__ src,
                           int B, int nsrc, int d, int roff, int nrows) {
    int idx = blockIdx.x * blockDim.x + threadIdx.x;
    if (idx >= B * nrows * d) return;
    int dd = idx % d; int i = (idx / d) % nrows; int b = idx / (d * nrows);
    dst[idx] += src[((size_t)b * nsrc + roff + i) * d + dd];
}

__global__ void k_ew_add(float* __restrict__ a, const float* __restrict__ b2, int n) {
    int idx = blockIdx.x * blockDim.x + threadIdx.x;
    if (idx < n) a[idx] += b2[idx];
}

__global__ void k_ew_max(const float* __restrict__ a, const float* __restrict__ b2,
                         float* __restrict__ o, int n) {
    int idx = blockIdx.x * blockDim.x + threadIdx.x;
    if (idx < n) o[idx] = fmaxf(a[idx], b2[idx]);
}

// last = [T_max | T_avg | S_max | S_avg | master] ; out = last @ ow + ob
__global__ void k_final_head(const float* __restrict__ tf, const float* __restrict__ sf,
                             const float* __restrict__ mf, const float* __restrict__ ow,
                             const float* __restrict__ ob, float* __restrict__ out,
                             int nT, int nS, int d) {
    __shared__ float last[160];
    int b = blockIdx.x, t = threadIdx.x;
    if (t < 160) {
        int seg = t / 32, dd = t % 32; float v = 0.f;
        if (seg == 0)      { for (int i = 0; i < nT; ++i) v = fmaxf(v, fabsf(tf[((size_t)b * nT + i) * d + dd])); }
        else if (seg == 1) { for (int i = 0; i < nT; ++i) v += tf[((size_t)b * nT + i) * d + dd]; v /= (float)nT; }
        else if (seg == 2) { for (int i = 0; i < nS; ++i) v = fmaxf(v, fabsf(sf[((size_t)b * nS + i) * d + dd])); }
        else if (seg == 3) { for (int i = 0; i < nS; ++i) v += sf[((size_t)b * nS + i) * d + dd]; v /= (float)nS; }
        else               { v = mf[(size_t)b * d + dd]; }
        last[t] = v;
    }
    __syncthreads();
    if (t < 2) {
        float acc = ob[t];
        for (int f = 0; f < 160; ++f) acc += last[f] * ow[f * 2 + t];
        out[b * 2 + t] = acc;
    }
}

// ---------------------------------------------------------------------------
// Host side
// ---------------------------------------------------------------------------
static inline int cdiv(int a, int b) { return (a + b - 1) / b; }

struct GatP  { const float *ap_w, *ap_b, *aw, *pwa_w, *pwa_b, *pwo_w, *pwo_b; };
struct HtrgP { const float *pt1_w, *pt1_b, *pt2_w, *pt2_b, *ap_w, *ap_b, *apM_w, *apM_b,
               *pwa_w, *pwa_b, *pwo_w, *pwo_b, *pwaM_w, *pwaM_b, *pwoM_w, *pwoM_b,
               *aw11, *aw22, *aw12, *awM; };
struct PoolP { const float *w, *b; };

static GatP load_gat(void* const* din, int base, bool sorted) {
    auto f = [&](int k) { return (const float*)din[base + k]; };
    GatP p;
    if (sorted) { p.ap_b = f(0); p.ap_w = f(1); p.aw = f(2); p.pwa_b = f(3); p.pwa_w = f(4); p.pwo_b = f(5); p.pwo_w = f(6); }
    else        { p.ap_w = f(0); p.ap_b = f(1); p.pwa_w = f(2); p.pwa_b = f(3); p.pwo_w = f(4); p.pwo_b = f(5); p.aw = f(6); }
    return p;
}
static HtrgP load_htrg(void* const* din, int base, bool sorted) {
    auto f = [&](int k) { return (const float*)din[base + k]; };
    HtrgP p;
    if (sorted) {
        p.apM_b = f(0);  p.apM_w = f(1);  p.ap_b  = f(2);  p.ap_w  = f(3);
        p.aw11  = f(4);  p.aw12  = f(5);  p.aw22  = f(6);  p.awM   = f(7);
        p.pt1_b = f(8);  p.pt1_w = f(9);  p.pt2_b = f(10); p.pt2_w = f(11);
        p.pwaM_b = f(12); p.pwaM_w = f(13); p.pwa_b = f(14); p.pwa_w = f(15);
        p.pwoM_b = f(16); p.pwoM_w = f(17); p.pwo_b = f(18); p.pwo_w = f(19);
    } else {
        p.pt1_w = f(0);  p.pt1_b = f(1);  p.pt2_w = f(2);  p.pt2_b = f(3);
        p.ap_w  = f(4);  p.ap_b  = f(5);  p.apM_w = f(6);  p.apM_b = f(7);
        p.pwa_w = f(8);  p.pwa_b = f(9);  p.pwo_w = f(10); p.pwo_b = f(11);
        p.pwaM_w = f(12); p.pwaM_b = f(13); p.pwoM_w = f(14); p.pwoM_b = f(15);
        p.aw11  = f(16); p.aw22  = f(17); p.aw12  = f(18); p.awM   = f(19);
    }
    return p;
}
static PoolP load_pool(void* const* din, int base, bool sorted) {
    PoolP p;
    if (sorted) { p.b = (const float*)din[base + 0]; p.w = (const float*)din[base + 1]; }
    else        { p.w = (const float*)din[base + 0]; p.b = (const float*)din[base + 1]; }
    return p;
}

extern "C" void kernel_launch(void* const* d_in, const int* in_sizes, int n_in,
                              void* d_out, int out_size, void* d_ws, size_t ws_size,
                              hipStream_t stream) {
    (void)n_in; (void)out_size; (void)ws_size;
    const int B = 8, D = 64, NS = 23, NT = 512;

    // Detect nested-dict flattening convention (insertion vs sorted-key order).
    bool sorted_order = (in_sizes[4] != 4096);

    const float* e       = (const float*)d_in[0];
    const float* pos     = (const float*)d_in[1];
    const float* master1 = (const float*)d_in[2];
    const float* master2 = (const float*)d_in[3];
    GatP  gS  = load_gat (d_in, 4,  sorted_order);
    GatP  gT  = load_gat (d_in, 11, sorted_order);
    HtrgP h11 = load_htrg(d_in, 18, sorted_order);
    HtrgP h12 = load_htrg(d_in, 38, sorted_order);
    HtrgP h21 = load_htrg(d_in, 58, sorted_order);
    HtrgP h22 = load_htrg(d_in, 78, sorted_order);
    PoolP pS  = load_pool(d_in, 98,  sorted_order);
    PoolP pT  = load_pool(d_in, 100, sorted_order);
    PoolP pS1 = load_pool(d_in, 102, sorted_order);
    PoolP pT1 = load_pool(d_in, 104, sorted_order);
    PoolP pS2 = load_pool(d_in, 106, sorted_order);
    PoolP pT2 = load_pool(d_in, 108, sorted_order);
    const float* out_w = (const float*)d_in[110];
    const float* out_b = (const float*)d_in[111];

    // Bump allocator over workspace.
    char* wsb = (char*)d_ws;
    size_t off = 0;
    auto alloc = [&](size_t nf) -> float* {
        float* p = (float*)(wsb + off);
        off += ((nf * sizeof(float) + 255) / 256) * 256;
        return p;
    };

    float* eS   = alloc((size_t)B * NS * D);
    float* eT   = alloc((size_t)B * NT * D);
    float* scS  = alloc((size_t)B * NS * NS);
    float* aggS = alloc((size_t)B * NS * D);
    float* hS   = alloc((size_t)B * NS * D);
    float* scT  = alloc((size_t)B * NT * NT);
    float* aggT = alloc((size_t)B * NT * D);
    float* hT   = alloc((size_t)B * NT * D);
    float* outS = alloc((size_t)B * 11 * 64);
    float* outT = alloc((size_t)B * 256 * 64);
    // branch scratch (reused by both branches, launched sequentially)
    float* xcat1 = alloc((size_t)B * 267 * 64);
    float* sc1   = alloc((size_t)B * 267 * 267);
    float* sM1   = alloc((size_t)B * 267);
    float* magg1 = alloc((size_t)B * 64);
    float* agg1  = alloc((size_t)B * 267 * 64);
    float* hH1   = alloc((size_t)B * 267 * 32);
    float* t1buf = alloc((size_t)B * 256 * 32);
    float* s1buf = alloc((size_t)B * 11 * 32);
    float* xcat2 = alloc((size_t)B * 133 * 32);
    float* sc2   = alloc((size_t)B * 133 * 133);
    float* sM2   = alloc((size_t)B * 133);
    float* magg2 = alloc((size_t)B * 32);
    float* agg2  = alloc((size_t)B * 133 * 32);
    float* hH2   = alloc((size_t)B * 133 * 32);
    float* maBuf = alloc((size_t)B * 32);
    // persistent branch outputs
    float* t1o = alloc((size_t)B * 128 * 32); float* s1o = alloc((size_t)B * 5 * 32); float* m1o = alloc((size_t)B * 32);
    float* t2o = alloc((size_t)B * 128 * 32); float* s2o = alloc((size_t)B * 5 * 32); float* m2o = alloc((size_t)B * 32);
    float* tF  = alloc((size_t)B * 128 * 32); float* sF  = alloc((size_t)B * 5 * 32); float* mF  = alloc((size_t)B * 32);

    // --- Stage 0 -----------------------------------------------------------
    k_reduce_eS<<<cdiv(B * NS * D, 256), 256, 0, stream>>>(e, pos, eS, B, D, NS, NT);
    k_reduce_eT<<<cdiv(B * NT * D, 256), 256, 0, stream>>>(e, eT, B, D, NS, NT);

    // --- GAT layers --------------------------------------------------------
    auto run_gat = [&](const float* x, int n, const GatP& P, float inv_temp,
                       float* sc, float* agg, float* hbuf) {
        hipLaunchKernelGGL((k_pair_att<64, 64>), dim3(cdiv(n, 16), n, B), dim3(32), 0, stream,
                           x, P.ap_w, P.ap_b, P.aw, P.aw, P.aw, sc, n, n, inv_temp);
        k_softmax_rows<<<B * n, 256, 0, stream>>>(sc, n);
        k_attn_agg<<<cdiv(B * n * 64, 256), 256, 0, stream>>>(sc, x, agg, B, n, n, 64);
        k_proj_selu<<<cdiv(B * n * 64, 256), 256, 0, stream>>>(agg, x, P.pwa_w, P.pwa_b,
                                                               P.pwo_w, P.pwo_b, hbuf, B * n, 64, 64);
    };
    run_gat(eS, NS, gS, 0.5f, scS, aggS, hS);
    k_graph_pool<<<B, 256, 0, stream>>>(hS, pS.w, pS.b, outS, NS, 64, 11);
    run_gat(eT, NT, gT, 0.5f, scT, aggT, hT);
    k_graph_pool<<<B, 256, 0, stream>>>(hT, pT.w, pT.b, outT, NT, 64, 256);

    // --- Heterogeneous layer ----------------------------------------------
    auto run_htrg = [&](const float* x1, int n1, const float* x2, int n2,
                        const float* master, int mstride, const HtrgP& P, int din,
                        float* xcat, float* sc, float* sM, float* magg,
                        float* agg, float* hbuf, float* mout) {
        int n = n1 + n2;
        const int dout = 32;
        k_linear<<<cdiv(B * n1 * din, 256), 256, 0, stream>>>(x1, P.pt1_w, P.pt1_b, xcat,
                                                              B, n1, din, din, n1 * din, n * din, 0);
        k_linear<<<cdiv(B * n2 * din, 256), 256, 0, stream>>>(x2, P.pt2_w, P.pt2_b, xcat,
                                                              B, n2, din, din, n2 * din, n * din, n1);
        dim3 g(cdiv(n, 16), n, B);
        if (din == 64)
            hipLaunchKernelGGL((k_pair_att<64, 32>), g, dim3(32), 0, stream,
                               xcat, P.ap_w, P.ap_b, P.aw11, P.aw22, P.aw12, sc, n, n1, 0.01f);
        else
            hipLaunchKernelGGL((k_pair_att<32, 32>), g, dim3(32), 0, stream,
                               xcat, P.ap_w, P.ap_b, P.aw11, P.aw22, P.aw12, sc, n, n1, 0.01f);
        k_softmax_rows<<<B * n, 256, 0, stream>>>(sc, n);
        // master branch
        k_master_att<<<cdiv(B * n, 256), 256, 0, stream>>>(xcat, master, mstride, P.apM_w, P.apM_b,
                                                           P.awM, sM, B, n, din, dout, 0.01f);
        k_softmax_rows<<<B, 256, 0, stream>>>(sM, n);
        k_attn_agg<<<cdiv(B * din, 256), 256, 0, stream>>>(sM, xcat, magg, B, 1, n, din);
        k_master_update<<<cdiv(B * dout, 256), 256, 0, stream>>>(magg, master, mstride,
                                                                 P.pwaM_w, P.pwaM_b, P.pwoM_w, P.pwoM_b,
                                                                 mout, B, din, dout);
        // node branch
        k_attn_agg<<<cdiv(B * n * din, 256), 256, 0, stream>>>(sc, xcat, agg, B, n, n, din);
        k_proj_selu<<<cdiv(B * n * dout, 256), 256, 0, stream>>>(agg, xcat, P.pwa_w, P.pwa_b,
                                                                 P.pwo_w, P.pwo_b, hbuf, B * n, din, dout);
    };

    auto run_branch = [&](const float* masterIn, const HtrgP& P1, const HtrgP& P2,
                          const PoolP& poolS, const PoolP& poolT,
                          float* tOut, float* sOut, float* mOut) {
        // htrg #1: T(256) + S(11), din 64 -> dout 32, master broadcast (stride 0)
        run_htrg(outT, 256, outS, 11, masterIn, 0, P1, 64,
                 xcat1, sc1, sM1, magg1, agg1, hH1, mOut);
        k_copy_rows<<<cdiv(B * 256 * 32, 256), 256, 0, stream>>>(hH1, t1buf, B, 267, 32, 0, 256);
        k_copy_rows<<<cdiv(B * 11 * 32, 256), 256, 0, stream>>>(hH1, s1buf, B, 267, 32, 256, 11);
        k_graph_pool<<<B, 256, 0, stream>>>(s1buf, poolS.w, poolS.b, sOut, 11, 32, 5);
        k_graph_pool<<<B, 256, 0, stream>>>(t1buf, poolT.w, poolT.b, tOut, 256, 32, 128);
        // htrg #2: T(128) + S(5), din 32 -> dout 32, master = mOut (stride 32)
        run_htrg(tOut, 128, sOut, 5, mOut, 32, P2, 32,
                 xcat2, sc2, sM2, magg2, agg2, hH2, maBuf);
        // residuals
        k_add_rows<<<cdiv(B * 128 * 32, 256), 256, 0, stream>>>(tOut, hH2, B, 133, 32, 0, 128);
        k_add_rows<<<cdiv(B * 5 * 32, 256), 256, 0, stream>>>(sOut, hH2, B, 133, 32, 128, 5);
        k_ew_add<<<cdiv(B * 32, 256), 256, 0, stream>>>(mOut, maBuf, B * 32);
    };

    run_branch(master1, h11, h12, pS1, pT1, t1o, s1o, m1o);
    run_branch(master2, h21, h22, pS2, pT2, t2o, s2o, m2o);

    // --- Combine + head ----------------------------------------------------
    k_ew_max<<<cdiv(B * 128 * 32, 256), 256, 0, stream>>>(t1o, t2o, tF, B * 128 * 32);
    k_ew_max<<<cdiv(B * 5 * 32, 256), 256, 0, stream>>>(s1o, s2o, sF, B * 5 * 32);
    k_ew_max<<<cdiv(B * 32, 256), 256, 0, stream>>>(m1o, m2o, mF, B * 32);
    k_final_head<<<B, 160, 0, stream>>>(tF, sF, mF, out_w, out_b, (float*)d_out, 128, 5, 32);
}
